// GraphConvLayer_90202903150661
// MI455X (gfx1250) — matrix operand order repaired and checked by reading, not compiled, not run on hardware
//
#include <hip/hip_runtime.h>

#define BN_EPS 1e-3f

typedef __attribute__((ext_vector_type(16))) __bf16 v16bf;
typedef __attribute__((ext_vector_type(8)))  float  v8f;
typedef __attribute__((ext_vector_type(4)))  unsigned int v4u;

union Frag16 { v16bf bf; v4u u[2]; };
union Pack8  { __bf16 h[8]; v4u u; };

__device__ __forceinline__ float gelu_exact(float x) {
  return 0.5f * x * (1.0f + erff(x * 0.70710678118654752f));
}

// tanh-form GELU as x*sigmoid(2z), via native v_exp_f32 (exp2) + v_rcp_f32.
// l = log2(e) * 1.5957691216 * x * (1 + 0.044715 x^2)
__device__ __forceinline__ float gelu_fast(float x) {
  float x2 = x * x;
  float l = x * __builtin_fmaf(x2, 0.10294322f, 2.3022078f);
  float u = __builtin_amdgcn_exp2f(l);
  float r = __builtin_amdgcn_rcpf(1.0f + u);
  return __builtin_fmaf(-x, r, x);  // large +x -> x, large -x -> 0
}

// ---------------------------------------------------------------------------
// Prep: fold BN scale into weights, pack into WMMA B-fragment layout (bf16).
// Packed layout: block = kTile(32)*8 + nTile(16); within block, lane = kHalf*16
// + nIn holds 16 K-ascending bf16 contiguously (32B per lane -> b128 loads).
// ---------------------------------------------------------------------------
__global__ void pack_kernel(const float* __restrict__ W, const float* __restrict__ g,
                            const float* __restrict__ v, int K,
                            __bf16* __restrict__ Wp) {
  int idx = blockIdx.x * blockDim.x + threadIdx.x;
  if (idx >= K * 128) return;
  int k = idx >> 7, n = idx & 127;
  float s = g[k] * rsqrtf(v[k] + BN_EPS);
  float w = W[idx] * s;
  int nT = n >> 4, nI = n & 15;
  int kT = k >> 5, kH = (k >> 4) & 1, kI = k & 15;
  int lane = kH * 16 + nI;
  int block = kT * 8 + nT;
  Wp[(size_t)block * 512 + lane * 16 + kI] = (__bf16)w;
}

__global__ void bias_fold_kernel(const float* __restrict__ W, const float* __restrict__ g,
                                 const float* __restrict__ b, const float* __restrict__ m,
                                 const float* __restrict__ v, const float* __restrict__ bias,
                                 int K, float* __restrict__ out) {
  int n = threadIdx.x;  // 128 threads
  float acc = bias[n];
  for (int k = 0; k < K; ++k) {
    float s = g[k] * rsqrtf(v[k] + BN_EPS);
    acc += (b[k] - m[k] * s) * W[k * 128 + n];
  }
  out[n] = acc;
}

// ---------------------------------------------------------------------------
// Core: one wave computes a 16-row strip x 128 cols via bf16 WMMA, f32 accum.
// A from LDS (row-major bf16, pitch in bf16 units), B from global packed wts.
// ---------------------------------------------------------------------------
template <int KT>
__device__ __forceinline__ void gemm_strip(const __bf16* __restrict__ A, int apitch,
                                           const __bf16* __restrict__ Wp,
                                           const float* __restrict__ bias,
                                           int lane, int rowBase, v8f* acc /*[8]*/) {
  const int lHalf = lane >> 4;
  const int lCol  = lane & 15;
  Frag16 af[KT];
  const __bf16* arow = A + (rowBase + lCol) * apitch;
#pragma unroll
  for (int kt = 0; kt < KT; ++kt) {
    af[kt].u[0] = *(const v4u*)(arow + kt * 32 + lHalf * 16);
    af[kt].u[1] = *(const v4u*)(arow + kt * 32 + lHalf * 16 + 8);
  }
#pragma unroll
  for (int nt = 0; nt < 8; ++nt) {
    v8f c = {0.f, 0.f, 0.f, 0.f, 0.f, 0.f, 0.f, 0.f};
#pragma unroll
    for (int kt = 0; kt < KT; ++kt) {
      Frag16 bfr;
      const v4u* bp = (const v4u*)(Wp + (size_t)(kt * 8 + nt) * 512 + lane * 16);
      bfr.u[0] = bp[0];
      bfr.u[1] = bp[1];
      c = __builtin_amdgcn_wmma_f32_16x16x32_bf16(false, af[kt].bf, false, bfr.bf,
                                                  (short)0, c, false, false);
    }
    float bn = bias[nt * 16 + lCol];
#pragma unroll
    for (int r = 0; r < 8; ++r) c[r] += bn;
    acc[nt] = c;
  }
}

// ---------------------------------------------------------------------------
// Edge pipeline: gather -> FFN1 -> FFN2 -> *ew -> atomic scatter to agg (L2).
// ---------------------------------------------------------------------------
__global__ __launch_bounds__(256) void edge_msg_kernel(
    const float* __restrict__ node_reps, const int* __restrict__ eDst,
    const int* __restrict__ eNbr, const float* __restrict__ ew,
    const __bf16* __restrict__ Wp1, const __bf16* __restrict__ Wp2,
    const float* __restrict__ b1f, const float* __restrict__ b2f,
    float* __restrict__ agg, int E) {
  constexpr int P = 136;  // bf16 pitch: 272B -> 4-bank shift/row, conflict-free strips
  __shared__ __align__(16) __bf16 Xs[128 * P];
  __shared__ __align__(16) __bf16 Hs[128 * P];
  __shared__ int dstS[128];
  __shared__ float ewS[128];

  const int tid = threadIdx.x;
  const int e0 = blockIdx.x * 128;

  if (tid < 128) {
    int e = e0 + tid;
    dstS[tid] = (e < E) ? eDst[e] : -1;
    ewS[tid] = (e < E) ? ew[e] : 0.f;
  }
  {
    int row = tid >> 1, half = tid & 1;
    int e = e0 + row;
    __bf16* dst = Xs + row * P + half * 64;
    if (e < E) {
      const float4* src = (const float4*)(node_reps + (size_t)eNbr[e] * 128 + half * 64);
#pragma unroll
      for (int j = 0; j < 8; ++j) {
        float4 a = src[2 * j], b = src[2 * j + 1];
        Pack8 p;
        p.h[0] = (__bf16)a.x; p.h[1] = (__bf16)a.y; p.h[2] = (__bf16)a.z; p.h[3] = (__bf16)a.w;
        p.h[4] = (__bf16)b.x; p.h[5] = (__bf16)b.y; p.h[6] = (__bf16)b.z; p.h[7] = (__bf16)b.w;
        *(v4u*)(dst + j * 8) = p.u;
      }
    } else {
      v4u z = {0u, 0u, 0u, 0u};
#pragma unroll
      for (int j = 0; j < 8; ++j) *(v4u*)(dst + j * 8) = z;
    }
  }
  __syncthreads();

  const int lane = tid & 31, wave = tid >> 5;
  const int rowBase = wave * 16;
  const int lHalf = lane >> 4, lCol = lane & 15;

  v8f acc[8];
  gemm_strip<4>(Xs, P, Wp1, b1f, lane, rowBase, acc);
#pragma unroll
  for (int nt = 0; nt < 8; ++nt)
#pragma unroll
    for (int r = 0; r < 8; ++r)
      Hs[(rowBase + lHalf * 8 + r) * P + nt * 16 + lCol] = (__bf16)gelu_fast(acc[nt][r]);
  // Each wave owns its 16-row strip of Hs: no workgroup barrier needed.
  gemm_strip<4>(Hs, P, Wp2, b2f, lane, rowBase, acc);

  int nodes[8];
  float ewr[8];
#pragma unroll
  for (int r = 0; r < 8; ++r) {
    int row = rowBase + lHalf * 8 + r;
    nodes[r] = dstS[row];
    ewr[r] = ewS[row];
  }
#pragma unroll
  for (int nt = 0; nt < 8; ++nt) {
    int col = nt * 16 + lCol;
#pragma unroll
    for (int r = 0; r < 8; ++r) {
      if (nodes[r] >= 0) {
        float val = gelu_fast(acc[nt][r]) * ewr[r];
        unsafeAtomicAdd(agg + (size_t)nodes[r] * 128 + col, val);  // L2-resident
      }
    }
  }
}

// ---------------------------------------------------------------------------
// Update: concat[x, agg] (K=256) -> FFN -> FFN -> L2 normalize -> out.
// ---------------------------------------------------------------------------
__global__ __launch_bounds__(256) void update_kernel(
    const float* __restrict__ node_reps, const float* __restrict__ agg,
    const __bf16* __restrict__ Wu1, const __bf16* __restrict__ Wu2,
    const float* __restrict__ bu1f, const float* __restrict__ bu2f,
    float* __restrict__ out, int N) {
  constexpr int P1 = 264;  // bf16 pitch for 256-wide input (528B rows)
  constexpr int P2 = 136;
  constexpr int PO = 132;  // f32 pitch, 528B rows: aliases Xs exactly
  __shared__ __align__(16) unsigned char smemA[128 * P1 * 2];
  __shared__ __align__(16) __bf16 Hs[128 * P2];
  __bf16* Xs = (__bf16*)smemA;
  float* Ot = (float*)smemA;

  const int tid = threadIdx.x;
  const int n0 = blockIdx.x * 128;

  {
    int row = tid >> 1, half = tid & 1;
    int n = n0 + row;
    __bf16* dst = Xs + row * P1 + half * 128;
    if (n < N) {
      const float4* src = (const float4*)((half ? agg : node_reps) + (size_t)n * 128);
#pragma unroll
      for (int j = 0; j < 16; ++j) {
        float4 a = src[2 * j], b = src[2 * j + 1];
        Pack8 p;
        p.h[0] = (__bf16)a.x; p.h[1] = (__bf16)a.y; p.h[2] = (__bf16)a.z; p.h[3] = (__bf16)a.w;
        p.h[4] = (__bf16)b.x; p.h[5] = (__bf16)b.y; p.h[6] = (__bf16)b.z; p.h[7] = (__bf16)b.w;
        *(v4u*)(dst + j * 8) = p.u;
      }
    } else {
      v4u z = {0u, 0u, 0u, 0u};
#pragma unroll
      for (int j = 0; j < 16; ++j) *(v4u*)(dst + j * 8) = z;
    }
  }
  __syncthreads();

  const int lane = tid & 31, wave = tid >> 5;
  const int rowBase = wave * 16;
  const int lHalf = lane >> 4, lCol = lane & 15;

  v8f acc[8];
  gemm_strip<8>(Xs, P1, Wu1, bu1f, lane, rowBase, acc);
#pragma unroll
  for (int nt = 0; nt < 8; ++nt)
#pragma unroll
    for (int r = 0; r < 8; ++r)
      Hs[(rowBase + lHalf * 8 + r) * P2 + nt * 16 + lCol] = (__bf16)gelu_fast(acc[nt][r]);
  gemm_strip<4>(Hs, P2, Wu2, bu2f, lane, rowBase, acc);
  // Ot aliases Xs, but each wave only touches its own strip rows (layer-1 A
  // reads fully complete before these writes, in program order within wave).
#pragma unroll
  for (int nt = 0; nt < 8; ++nt)
#pragma unroll
    for (int r = 0; r < 8; ++r)
      Ot[(rowBase + lHalf * 8 + r) * PO + nt * 16 + lCol] = gelu_exact(acc[nt][r]);
  __syncthreads();

  if (tid < 128) {
    int n = n0 + tid;
    if (n < N) {
      const float4* orow = (const float4*)(Ot + tid * PO);
      float ss = 0.f;
#pragma unroll
      for (int j = 0; j < 32; ++j) {
        float4 f = orow[j];
        ss += f.x * f.x + f.y * f.y + f.z * f.z + f.w * f.w;
      }
      float rn = rsqrtf(fmaxf(ss, 1e-12f));
      float4* dst = (float4*)(out + (size_t)n * 128);
#pragma unroll
      for (int j = 0; j < 32; ++j) {
        float4 f = orow[j];
        dst[j] = make_float4(f.x * rn, f.y * rn, f.z * rn, f.w * rn);
      }
    }
  }
}

// ---------------------------------------------------------------------------
extern "C" void kernel_launch(void* const* d_in, const int* in_sizes, int n_in,
                              void* d_out, int out_size, void* d_ws, size_t ws_size,
                              hipStream_t stream) {
  (void)n_in; (void)out_size; (void)ws_size;
  const float* node_reps = (const float*)d_in[0];
  const int* edges = (const int*)d_in[1];
  const float* ew = (const float*)d_in[2];
  const int E = in_sizes[2];
  const int N = in_sizes[0] / 128;

  const float *g1 = (const float*)d_in[3],  *be1 = (const float*)d_in[4];
  const float *m1 = (const float*)d_in[5],  *v1  = (const float*)d_in[6];
  const float *W1 = (const float*)d_in[7],  *bi1 = (const float*)d_in[8];
  const float *g2 = (const float*)d_in[9],  *be2 = (const float*)d_in[10];
  const float *m2 = (const float*)d_in[11], *v2  = (const float*)d_in[12];
  const float *W2 = (const float*)d_in[13], *bi2 = (const float*)d_in[14];
  const float *gu1 = (const float*)d_in[15], *beu1 = (const float*)d_in[16];
  const float *mu1 = (const float*)d_in[17], *vu1  = (const float*)d_in[18];
  const float *Wu1w = (const float*)d_in[19], *biu1 = (const float*)d_in[20];
  const float *gu2 = (const float*)d_in[21], *beu2 = (const float*)d_in[22];
  const float *mu2 = (const float*)d_in[23], *vu2  = (const float*)d_in[24];
  const float *Wu2w = (const float*)d_in[25], *biu2 = (const float*)d_in[26];

  char* ws = (char*)d_ws;
  __bf16* Wp1 = (__bf16*)(ws);             // 32768 B
  __bf16* Wp2 = (__bf16*)(ws + 32768);     // 32768 B
  __bf16* Wu1 = (__bf16*)(ws + 65536);     // 65536 B
  __bf16* Wu2 = (__bf16*)(ws + 131072);    // 32768 B
  float* fb1 = (float*)(ws + 163840);
  float* fb2 = (float*)(ws + 164352);
  float* fu1 = (float*)(ws + 164864);
  float* fu2 = (float*)(ws + 165376);
  float* agg = (float*)(ws + 165888);      // N*128 f32

  hipMemsetAsync(agg, 0, (size_t)N * 128 * sizeof(float), stream);

  pack_kernel<<<(128 * 128) / 256, 256, 0, stream>>>(W1, g1, v1, 128, Wp1);
  pack_kernel<<<(128 * 128) / 256, 256, 0, stream>>>(W2, g2, v2, 128, Wp2);
  pack_kernel<<<(256 * 128) / 256, 256, 0, stream>>>(Wu1w, gu1, vu1, 256, Wu1);
  pack_kernel<<<(128 * 128) / 256, 256, 0, stream>>>(Wu2w, gu2, vu2, 128, Wu2);
  bias_fold_kernel<<<1, 128, 0, stream>>>(W1, g1, be1, m1, v1, bi1, 128, fb1);
  bias_fold_kernel<<<1, 128, 0, stream>>>(W2, g2, be2, m2, v2, bi2, 128, fb2);
  bias_fold_kernel<<<1, 128, 0, stream>>>(Wu1w, gu1, beu1, mu1, vu1, biu1, 256, fu1);
  bias_fold_kernel<<<1, 128, 0, stream>>>(Wu2w, gu2, beu2, mu2, vu2, biu2, 128, fu2);

  edge_msg_kernel<<<(E + 127) / 128, 256, 0, stream>>>(
      node_reps, edges, edges + E, ew, Wp1, Wp2, fb1, fb2, agg, E);

  update_kernel<<<(N + 127) / 128, 256, 0, stream>>>(
      node_reps, agg, Wu1, Wu2, fu1, fu2, (float*)d_out, N);
}
